// CL_Attention_19636590478163
// MI455X (gfx1250) — matrix operand order repaired and checked by reading, not compiled, not run on hardware
//
#include <hip/hip_runtime.h>
#include <hip/hip_bf16.h>

typedef __attribute__((ext_vector_type(16))) _Float16 v16h;
typedef __attribute__((ext_vector_type(8)))  float    v8f;

#define BATCH 8
#define CC    64
#define HH    112
#define WW    112
#define HW    (HH * WW)   // 12544
#define KN    9

// ---------------------------------------------------------------------------
// Kernel A: key_feature[b,o,p] = sum_c w_refer[o,c] * key_layer[b,c,p] + b_refer[o]
// GEMM per batch: [64x64] x [64x12544], f16 WMMA, f32 accumulate.
// Block = 256 threads = 8 wave32s; each wave owns one 16(o) x 16(p) tile;
// block covers all 64 o-channels x 32 pixels.
// ---------------------------------------------------------------------------
__global__ __launch_bounds__(256)
void kf_gemm_wmma(const float* __restrict__ x,    // key_layer  [B,64,HW]
                  const float* __restrict__ w,    // w_refer    [64,64]
                  const float* __restrict__ bias, // b_refer    [64]
                  float* __restrict__ y)          // key_feature[B,64,HW]
{
    const int blk  = blockIdx.x;                 // B * (HW/32) blocks
    const int b    = blk / (HW / 32);
    const int p0   = (blk % (HW / 32)) * 32;
    const int wave = threadIdx.x >> 5;           // 0..7
    const int lane = threadIdx.x & 31;
    const int otile = (wave & 3) * 16;           // o base: 0,16,32,48
    const int ptile = p0 + (wave >> 2) * 16;     // p base: p0, p0+16

    const float* X = x + (size_t)b * CC * HW;
    float*       Y = y + (size_t)b * CC * HW;

    // WMMA fragment lane mapping (ISA 7.12.2, wave32):
    const int row  = lane & 15;            // A: matrix row M
    const int kbA  = (lane >> 4) << 3;     // A: K offset 0 / 8
    const int col  = lane & 15;            // B: matrix col N
    const int kbB  = (lane >> 4) << 4;     // B: K offset 0 / 16
    const int mrow = (lane >> 4) << 3;     // C/D: M offset 0 / 8

    // init accumulator with bias (bias depends only on M = output channel)
    v8f acc;
#pragma unroll
    for (int r = 0; r < 8; ++r)
        acc[r] = bias[otile + mrow + r];

#pragma unroll
    for (int kk = 0; kk < 2; ++kk) {       // K = 64 -> two x32 steps
        const int cbase = kk * 32;
        v16h af, bf;
#pragma unroll
        for (int i = 0; i < 8; ++i) {
            // A (16x32 f16): VGPR i holds K pair; K = kbA + (i<4?0:16) + 2*(i&3)
            const int ka = cbase + kbA + ((i >> 2) << 4) + ((i & 3) << 1);
            af[2 * i]     = (_Float16)w[(otile + row) * CC + ka];
            af[2 * i + 1] = (_Float16)w[(otile + row) * CC + ka + 1];
            // B (32x16 f16): lane col = N, K = kbA + (i<4?0:16) + 2*(i&3) (+1), same as A
            const int kb = cbase + kbA + ((i >> 2) << 4) + ((i & 3) << 1);
            bf[2 * i]     = (_Float16)X[(size_t)kb * HW + ptile + col];
            bf[2 * i + 1] = (_Float16)X[(size_t)(kb + 1) * HW + ptile + col];
        }
        acc = __builtin_amdgcn_wmma_f32_16x16x32_f16(
            /*neg_a=*/false, af, /*neg_b=*/false, bf,
            /*c_mod=*/(short)0, acc, /*reuse_a=*/false, /*reuse_b=*/false);
    }

#pragma unroll
    for (int r = 0; r < 8; ++r) {
        const int o = otile + mrow + r;
        Y[(size_t)o * HW + ptile + col] = acc[r];
    }
}

// ---------------------------------------------------------------------------
// Kernel B (fused): per pixel -> attn logits (9) + offsets (channels 0..9),
// softmax, 9x bilinear descriptor precompute, then stream 64 channels of
// key_feature with 36 register-resident (index, weight) gathers per channel.
// ---------------------------------------------------------------------------
__global__ __launch_bounds__(256)
void deform_attn_fused(const float* __restrict__ q,      // query   [B,64,HW]
                       const float* __restrict__ kf,     // keyfeat [B,64,HW]
                       const float* __restrict__ w_attn, // [9,64]
                       const float* __restrict__ b_attn, // [9]
                       const float* __restrict__ w_off,  // [18,64]
                       const float* __restrict__ b_off,  // [18]
                       float* __restrict__ out)          // [B,64,HW]
{
    // Stage transposed weights: j = 0..8 -> w_attn rows, j = 9..18 -> w_off rows 0..9
    __shared__ float wT[CC][20];
    __shared__ float bb[19];
    for (int t = threadIdx.x; t < CC * 19; t += blockDim.x) {
        const int c = t / 19, j = t % 19;
        wT[c][j] = (j < KN) ? w_attn[j * CC + c] : w_off[(j - KN) * CC + c];
    }
    if (threadIdx.x < 19)
        bb[threadIdx.x] = (threadIdx.x < KN) ? b_attn[threadIdx.x]
                                             : b_off[threadIdx.x - KN];
    __syncthreads();

    const int tid = blockIdx.x * blockDim.x + threadIdx.x;   // 0..B*HW-1
    const int b   = tid / HW;
    const int p   = tid % HW;
    const int yi  = p / WW;
    const int xi  = p % WW;

    // 19 dot products over the 64 query channels (coalesced loads, LDS broadcast)
    float acc[19];
#pragma unroll
    for (int j = 0; j < 19; ++j) acc[j] = bb[j];
    const float* qb = q + (size_t)b * CC * HW + p;
    for (int c = 0; c < CC; ++c) {
        const float qv = qb[(size_t)c * HW];
#pragma unroll
        for (int j = 0; j < 19; ++j) acc[j] = fmaf(qv, wT[c][j], acc[j]);
    }

    // softmax over acc[0..8]
    float m = acc[0];
#pragma unroll
    for (int j = 1; j < KN; ++j) m = fmaxf(m, acc[j]);
    float s = 0.f;
#pragma unroll
    for (int j = 0; j < KN; ++j) { acc[j] = __expf(acc[j] - m); s += acc[j]; }
    const float inv = 1.f / s;

    // per-key bilinear descriptors (attn weight folded in); all constant-indexed
    int   sidx[4 * KN];
    float sw[4 * KN];
#pragma unroll
    for (int k = 0; k < KN; ++k) {
        const float a  = acc[k] * inv;
        const float dy = acc[KN + k];        // offset channel k     (bias included)
        const float dx = acc[KN + 1 + k];    // offset channel k + 1 (overlap quirk)
        const float py = dy + (float)yi;
        const float px = dx + (float)xi;
        const float y0f = floorf(py), x0f = floorf(px);
        const float wy1 = py - y0f,   wx1 = px - x0f;
        const float wy0 = 1.f - wy1,  wx0 = 1.f - wx1;
        const int iy0 = (int)y0f, ix0 = (int)x0f;
        const int iy1 = iy0 + 1,  ix1 = ix0 + 1;
        const bool vy0 = (iy0 >= 0) && (iy0 < HH);
        const bool vy1 = (iy1 >= 0) && (iy1 < HH);
        const bool vx0 = (ix0 >= 0) && (ix0 < WW);
        const bool vx1 = (ix1 >= 0) && (ix1 < WW);
        const int cy0 = iy0 < 0 ? 0 : (iy0 > HH - 1 ? HH - 1 : iy0);
        const int cy1 = iy1 < 0 ? 0 : (iy1 > HH - 1 ? HH - 1 : iy1);
        const int cx0 = ix0 < 0 ? 0 : (ix0 > WW - 1 ? WW - 1 : ix0);
        const int cx1 = ix1 < 0 ? 0 : (ix1 > WW - 1 ? WW - 1 : ix1);
        sidx[4 * k + 0] = cy0 * WW + cx0;  sw[4 * k + 0] = (vy0 && vx0) ? a * wy0 * wx0 : 0.f;
        sidx[4 * k + 1] = cy0 * WW + cx1;  sw[4 * k + 1] = (vy0 && vx1) ? a * wy0 * wx1 : 0.f;
        sidx[4 * k + 2] = cy1 * WW + cx0;  sw[4 * k + 2] = (vy1 && vx0) ? a * wy1 * wx0 : 0.f;
        sidx[4 * k + 3] = cy1 * WW + cx1;  sw[4 * k + 3] = (vy1 && vx1) ? a * wy1 * wx1 : 0.f;
    }

    // stream all 64 channels: 36 gathers + FMAs each, then one coalesced store
    const float* kbase = kf + (size_t)b * CC * HW;
    float*       obase = out + (size_t)b * CC * HW + p;
    for (int c = 0; c < CC; ++c) {
        const float* plane = kbase + (size_t)c * HW;
        float sv = 0.f;
#pragma unroll
        for (int j = 0; j < 4 * KN; ++j)
            sv = fmaf(sw[j], plane[sidx[j]], sv);
        obase[(size_t)c * HW] = sv;
    }
}

// ---------------------------------------------------------------------------
extern "C" void kernel_launch(void* const* d_in, const int* in_sizes, int n_in,
                              void* d_out, int out_size, void* d_ws, size_t ws_size,
                              hipStream_t stream) {
    const float* query  = (const float*)d_in[0];
    const float* keylay = (const float*)d_in[1];
    const float* w_ref  = (const float*)d_in[2];
    const float* b_ref  = (const float*)d_in[3];
    const float* w_attn = (const float*)d_in[4];
    const float* b_attn = (const float*)d_in[5];
    const float* w_off  = (const float*)d_in[6];
    const float* b_off  = (const float*)d_in[7];
    float* out   = (float*)d_out;
    float* kfeat = (float*)d_ws;   // B*C*HW f32 = ~25.7 MB scratch

    // Kernel A: 8 batches x (12544/32) pixel strips, 256 threads (8 waves)
    dim3 grdA(BATCH * (HW / 32));
    kf_gemm_wmma<<<grdA, dim3(256), 0, stream>>>(keylay, w_ref, b_ref, kfeat);

    // Kernel B: one thread per pixel
    dim3 grdB((BATCH * HW) / 256);   // 392 blocks
    deform_attn_fused<<<grdB, dim3(256), 0, stream>>>(query, kfeat, w_attn, b_attn,
                                                      w_off, b_off, out);
}
